// GRNGOB_58609123721580
// MI455X (gfx1250) — compile-verified, matrix operand order checked by reading, not run on hardware
//
#include <hip/hip_runtime.h>
#include <stdint.h>

// ---------------- constants ----------------
#define Bn 8
#define Sn 64
#define Wn 128
#define HSn 512
#define Ln 128

typedef __attribute__((ext_vector_type(16))) __bf16   v16bf;
typedef __attribute__((ext_vector_type(8)))  float    v8f;
typedef __attribute__((ext_vector_type(8)))  uint32_t v8u;
typedef __attribute__((ext_vector_type(4)))  uint32_t v4u;

union FragU { v8u u; v16bf b; };

__device__ __forceinline__ uint16_t f2bf(float f) {
  union { float f; uint32_t u; } v; v.f = f;
  uint32_t r = v.u + 0x7FFFu + ((v.u >> 16) & 1u);
  return (uint16_t)(r >> 16);
}
__device__ __forceinline__ uint32_t pack2bf(float a, float b) {
  return (uint32_t)f2bf(a) | ((uint32_t)f2bf(b) << 16);
}
__device__ __forceinline__ float sigmoidf_(float x) { return 1.0f / (1.0f + __expf(-x)); }

// Async global->LDS copy, 16B per lane, tracked by ASYNCcnt (CDNA5).
__device__ __forceinline__ void async_b128(uint32_t lds_off, unsigned long long gaddr) {
  asm volatile("global_load_async_to_lds_b128 %0, %1, off"
               :: "v"(lds_off), "v"(gaddr) : "memory");
}
__device__ __forceinline__ void wait_async0() {
  asm volatile("s_wait_asynccnt 0x0" ::: "memory");
}

// A-fragment (16x32 bf16) from a row-major LDS tile (row stride 256B = 64 dwords).
// lane<16: M=lane, K groups {0-7},{16-23}; lane>=16: M=lane-16, K groups {8-15},{24-31}.
__device__ __forceinline__ FragU load_afrag(const uint32_t* smem, int mt, int kt, int lane) {
  int m = lane & 15, hi = lane >> 4;
  const uint32_t* rp = smem + ((mt << 4) + m) * 64 + (kt << 4) + (hi << 2);
  v4u lo = *(const v4u*)rp;
  v4u hh = *(const v4u*)(rp + 8);
  FragU a;
  a.u = __builtin_shufflevector(lo, hh, 0, 1, 2, 3, 4, 5, 6, 7);
  return a;
}

// ---------------- weight packing into B-fragment order ----------------
__global__ void k_pack_w(const float* __restrict__ W, uint32_t* __restrict__ F, int K, int N) {
  int idx = blockIdx.x * blockDim.x + threadIdx.x;
  int nTiles = N >> 4;
  int total = (K >> 5) * nTiles * 256;
  if (idx >= total) return;
  int v    = idx & 7;
  int lane = (idx >> 3) & 31;
  int tile = idx >> 8;
  int nt   = tile % nTiles;
  int ktg  = tile / nTiles;
  int n  = (nt << 4) + (lane & 15);
  int hi = lane >> 4;
  int kk = ((v & 3) << 1) + ((v & 4) << 2) + (hi << 3);
  int Kk = (ktg << 5) + kk;
  F[idx] = pack2bf(W[(size_t)Kk * N + n], W[(size_t)(Kk + 1) * N + n]);
}

// ---------------- generic bf16 WMMA GEMM: C = A*W + bias (fp32 acc) -------
// block 256 (8 waves), tile 128x64, K chunked by 128 (K % 128 == 0).
__global__ void __launch_bounds__(256)
k_gemm_bf16(const uint16_t* __restrict__ A, int lda,
            const uint32_t* __restrict__ F, int nTilesW,
            const float* __restrict__ bias,
            float* __restrict__ C, int ldc, int K) {
  __shared__ uint32_t smem[8192];  // 128 rows x 256B, row-major
  int tid = threadIdx.x, wid = tid >> 5, lane = tid & 31;
  int bm = blockIdx.x, bnb = blockIdx.y;
  uint32_t ldsb = (uint32_t)(size_t)&smem[0];
  v8f acc[4] = {};
  int kchunks = K >> 7;
  for (int kc = 0; kc < kchunks; ++kc) {
    for (int it = 0; it < 8; ++it) {
      int t16 = ((it << 8) + tid) << 4;     // byte index in 32KB tile
      int r = t16 >> 8, cb = t16 & 255;
      unsigned long long g = (unsigned long long)(size_t)A +
          ((size_t)(bm * 128 + r) * lda + ((size_t)kc << 7)) * 2 + (unsigned)cb;
      async_b128(ldsb + (uint32_t)t16, g);
    }
    wait_async0();
    __syncthreads();
    for (int kt = 0; kt < 4; ++kt) {
      FragU a = load_afrag(smem, wid, kt, lane);
      int ktg = (kc << 2) + kt;
      for (int nt = 0; nt < 4; ++nt) {
        FragU b;
        b.u = *(const v8u*)(F + (((size_t)ktg * nTilesW + (bnb << 2) + nt) << 8) + (lane << 3));
        acc[nt] = __builtin_amdgcn_wmma_f32_16x16x32_bf16(
            false, a.b, false, b.b, (short)0, acc[nt], false, false);
      }
    }
    __syncthreads();
  }
  int nl = lane & 15, rowoff = (lane >> 4) << 3;
  for (int nt = 0; nt < 4; ++nt) {
    int n = (bnb << 6) + (nt << 4) + nl;
    float bv = bias ? bias[n] : 0.0f;
    int row0 = bm * 128 + (wid << 4) + rowoff;
    for (int i = 0; i < 8; ++i)
      C[(size_t)(row0 + i) * ldc + n] = acc[nt][i] + bv;
  }
}

// ---------------- fused gate: sentence <- entity ----------------
// grid (B*S, 10). M=128 (w rows), N=64 cols/block, K=1152 (9 chunks of 128).
__global__ void __launch_bounds__(256)
k_gate_s(const uint16_t* __restrict__ sh_bf, const uint16_t* __restrict__ eh_bf,
         const uint16_t* __restrict__ lab_bf, const float* __restrict__ eh_f,
         const float* __restrict__ edge_emb, const int* __restrict__ matrix,
         const uint32_t* __restrict__ F, const float* __restrict__ bias,
         float* __restrict__ out) {
  __shared__ uint32_t smem[8192];
  __shared__ float sbuf[64];
  int tid = threadIdx.x, wid = tid >> 5, lane = tid & 31;
  int bs = blockIdx.x;          // b*S + s
  int b = bs >> 6;
  int bnb = blockIdx.y;
  uint32_t ldsb = (uint32_t)(size_t)&smem[0];
  v8f acc[4] = {};
  if (tid < 64) sbuf[tid] = 0.0f;
  for (int kc = 0; kc < 9; ++kc) {
    for (int it = 0; it < 8; ++it) {
      int t16 = ((it << 8) + tid) << 4;
      int r = t16 >> 8, cb = t16 & 255;
      int K0b = kc * 256 + cb;            // byte offset within 2304-byte logical row
      unsigned long long g;
      if (K0b < 1024)       // s_h[b,s], broadcast to all rows
        g = (unsigned long long)(size_t)sh_bf + (size_t)bs * 1024 + (unsigned)K0b;
      else if (K0b < 2048)  // e_h[b,r]
        g = (unsigned long long)(size_t)eh_bf + (size_t)(b * Wn + r) * 1024 + (unsigned)(K0b - 1024);
      else                  // label[b,s,r]
        g = (unsigned long long)(size_t)lab_bf + ((size_t)bs * Wn + r) * 256 + (unsigned)(K0b - 2048);
      async_b128(ldsb + (uint32_t)t16, g);
    }
    wait_async0();
    __syncthreads();
    for (int kt = 0; kt < 4; ++kt) {
      FragU a = load_afrag(smem, wid, kt, lane);
      int ktg = kc * 4 + kt;
      for (int nt = 0; nt < 4; ++nt) {
        FragU bf;
        bf.u = *(const v8u*)(F + (((size_t)ktg * 40 + bnb * 4 + nt) << 8) + (lane << 3));
        acc[nt] = __builtin_amdgcn_wmma_f32_16x16x32_bf16(
            false, a.b, false, bf.b, (short)0, acc[nt], false, false);
      }
    }
    __syncthreads();
  }
  int nl = lane & 15, rowoff = (lane >> 4) << 3;
  for (int nt = 0; nt < 4; ++nt) {
    int n = bnb * 64 + nt * 16 + nl;
    float bv = bias[n];
    float partial = 0.0f;
    for (int i = 0; i < 8; ++i) {
      int w = (wid << 4) + rowoff + i;
      int mv = matrix[(size_t)bs * Wn + w];
      float g = sigmoidf_(acc[nt][i] + bv);
      float ehe = (n < 512) ? eh_f[(size_t)(b * Wn + w) * 512 + n]
                            : edge_emb[mv * Ln + (n - 512)];
      partial += g * ehe * (mv != 0 ? 1.0f : 0.0f);
    }
    partial += __shfl_xor(partial, 16, 32);
    if (lane < 16) atomicAdd(&sbuf[nt * 16 + nl], partial);
  }
  __syncthreads();
  if (tid < 64) out[(size_t)bs * 640 + bnb * 64 + tid] = sbuf[tid];
}

// ---------------- fused gate: entity <- sentence ----------------
// grid (B*W, 10). M=64 (s rows): 4 m-tiles, each handled by 2 waves (2 n-tiles each).
__global__ void __launch_bounds__(256)
k_gate_e(const uint16_t* __restrict__ sh_bf, const uint16_t* __restrict__ eh_bf,
         const uint16_t* __restrict__ lab_bf, const float* __restrict__ sh_f,
         const float* __restrict__ edge_emb, const int* __restrict__ matrix,
         const uint32_t* __restrict__ F, const float* __restrict__ bias,
         float* __restrict__ out) {
  __shared__ uint32_t smem[4096];  // 64 rows x 256B
  __shared__ float sbuf[64];
  int tid = threadIdx.x, wid = tid >> 5, lane = tid & 31;
  int bw = blockIdx.x;          // b*W + w
  int b = bw >> 7, w = bw & 127;
  int bnb = blockIdx.y;
  int mt = wid & 3, ng = wid >> 2;
  uint32_t ldsb = (uint32_t)(size_t)&smem[0];
  v8f acc[2] = {};
  if (tid < 64) sbuf[tid] = 0.0f;
  for (int kc = 0; kc < 9; ++kc) {
    for (int it = 0; it < 4; ++it) {
      int t16 = ((it << 8) + tid) << 4;
      int r = t16 >> 8, cb = t16 & 255;
      int K0b = kc * 256 + cb;
      unsigned long long g;
      if (K0b < 1024)       // e_h[b,w], broadcast to all rows
        g = (unsigned long long)(size_t)eh_bf + (size_t)bw * 1024 + (unsigned)K0b;
      else if (K0b < 2048)  // s_h[b,r]
        g = (unsigned long long)(size_t)sh_bf + (size_t)(b * Sn + r) * 1024 + (unsigned)(K0b - 1024);
      else                  // label[b,r,w]
        g = (unsigned long long)(size_t)lab_bf + ((size_t)(b * Sn + r) * Wn + w) * 256 + (unsigned)(K0b - 2048);
      async_b128(ldsb + (uint32_t)t16, g);
    }
    wait_async0();
    __syncthreads();
    for (int kt = 0; kt < 4; ++kt) {
      FragU a = load_afrag(smem, mt, kt, lane);
      int ktg = kc * 4 + kt;
      for (int j = 0; j < 2; ++j) {
        int nt = ng * 2 + j;
        FragU bf;
        bf.u = *(const v8u*)(F + (((size_t)ktg * 40 + bnb * 4 + nt) << 8) + (lane << 3));
        acc[j] = __builtin_amdgcn_wmma_f32_16x16x32_bf16(
            false, a.b, false, bf.b, (short)0, acc[j], false, false);
      }
    }
    __syncthreads();
  }
  int nl = lane & 15, rowoff = (lane >> 4) << 3;
  for (int j = 0; j < 2; ++j) {
    int nt = ng * 2 + j;
    int n = bnb * 64 + nt * 16 + nl;
    float bv = bias[n];
    float partial = 0.0f;
    for (int i = 0; i < 8; ++i) {
      int s = (mt << 4) + rowoff + i;
      int mv = matrix[(size_t)(b * Sn + s) * Wn + w];
      float g = sigmoidf_(acc[j][i] + bv);
      float she = (n < 512) ? sh_f[(size_t)(b * Sn + s) * 512 + n]
                            : edge_emb[mv * Ln + (n - 512)];
      partial += g * she * (mv != 0 ? 1.0f : 0.0f);
    }
    partial += __shfl_xor(partial, 16, 32);
    if (lane < 16) atomicAdd(&sbuf[nt * 16 + nl], partial);
  }
  __syncthreads();
  if (tid < 64) out[(size_t)bw * 640 + bnb * 64 + tid] = sbuf[tid];
}

// ---------------- small elementwise / helper kernels ----------------
__global__ void k_zero(float* p, int n) {
  int i = blockIdx.x * blockDim.x + threadIdx.x;
  if (i < n) p[i] = 0.0f;
}
__global__ void k_f2bf(const float* __restrict__ s, uint16_t* __restrict__ d, int n) {
  int i = blockIdx.x * blockDim.x + threadIdx.x;
  if (i < n) d[i] = f2bf(s[i]);
}
__global__ void k_label_bf(const int* __restrict__ matrix, const float* __restrict__ edge_emb,
                           uint16_t* __restrict__ lab) {
  size_t idx = (size_t)blockIdx.x * blockDim.x + threadIdx.x;
  size_t total = (size_t)Bn * Sn * Wn * Ln;
  if (idx >= total) return;
  int l = (int)(idx & (Ln - 1));
  size_t bsw = idx >> 7;
  int mv = matrix[bsw];
  lab[idx] = f2bf(edge_emb[mv * Ln + l]);
}
__global__ void k_sns(const float* __restrict__ ord, const float* __restrict__ sh,
                      float* __restrict__ out) {
  int idx = blockIdx.x * blockDim.x + threadIdx.x;
  if (idx >= Bn * Sn * HSn) return;
  int h = idx & 511;
  int bi = idx >> 9;
  int b = bi >> 6;
  const float* o = ord + (size_t)bi * Sn;
  const float* sp = sh + (size_t)(b * Sn) * HSn + h;
  float acc = 0.0f;
  for (int j = 0; j < Sn; ++j) acc += o[j] * sp[(size_t)j * HSn];
  out[idx] = acc;
}
__global__ void k_concat_s(const float* __restrict__ sent, const float* __restrict__ sns,
                           const float* __restrict__ sne, const float* __restrict__ gh,
                           const float* __restrict__ sh, uint16_t* __restrict__ xh) {
  int idx = blockIdx.x * blockDim.x + threadIdx.x;
  if (idx >= 512 * 2688) return;
  int c = idx % 2688, row = idx / 2688, b = row >> 6;
  float v;
  if (c < 512)       v = sent[(size_t)row * 512 + c];
  else if (c < 1024) v = sns[(size_t)row * 512 + (c - 512)];
  else if (c < 1664) v = sne[(size_t)row * 640 + (c - 1024)];
  else if (c < 2176) v = gh[(size_t)b * 512 + (c - 1664)];
  else               v = sh[(size_t)row * 512 + (c - 2176)];
  xh[idx] = f2bf(v);
}
__global__ void k_concat_e(const float* __restrict__ word, const float* __restrict__ ens,
                           const float* __restrict__ gh, const float* __restrict__ eh,
                           uint16_t* __restrict__ xh) {
  int idx = blockIdx.x * blockDim.x + threadIdx.x;
  if (idx >= 1024 * 2176) return;
  int c = idx % 2176, row = idx / 2176, b = row >> 7;
  float v;
  if (c < 512)       v = word[(size_t)row * 512 + c];
  else if (c < 1152) v = ens[(size_t)row * 640 + (c - 512)];
  else if (c < 1664) v = gh[(size_t)b * 512 + (c - 1152)];
  else               v = eh[(size_t)row * 512 + (c - 1664)];
  xh[idx] = f2bf(v);
}
__global__ void k_gru_point(const float* __restrict__ r_, const float* __restrict__ z_,
                            const float* __restrict__ c_, const float* __restrict__ ul,
                            const float* __restrict__ h, const int* __restrict__ mask,
                            float* __restrict__ out, int rows) {
  int idx = blockIdx.x * blockDim.x + threadIdx.x;
  if (idx >= rows * 512) return;
  int row = idx >> 9;
  float r = sigmoidf_(r_[idx]);
  float z = sigmoidf_(z_[idx]);
  float u = tanhf(c_[idx] + r * ul[idx]);
  float nh = z * h[idx] + (1.0f - z) * u;
  out[idx] = nh * (mask[row] ? 1.0f : 0.0f);
}
__global__ void k_means(const float* __restrict__ sh, const float* __restrict__ eh,
                        const int* __restrict__ smask, const int* __restrict__ wmask,
                        float* __restrict__ shm, float* __restrict__ ehm) {
  int idx = blockIdx.x * blockDim.x + threadIdx.x;
  if (idx >= Bn * 512) return;
  int b = idx >> 9, h = idx & 511;
  float ssum = 0.0f, scnt = 0.0f;
  for (int s = 0; s < Sn; ++s) ssum += sh[(size_t)(b * Sn + s) * 512 + h];
  for (int s = 0; s < Sn; ++s) scnt += (float)smask[b * Sn + s];
  shm[idx] = ssum / scnt;
  float esum = 0.0f, ecnt = 0.0f;
  for (int w = 0; w < Wn; ++w) esum += eh[(size_t)(b * Wn + w) * 512 + h];
  for (int w = 0; w < Wn; ++w) ecnt += (float)wmask[b * Wn + w];
  ehm[idx] = esum / (ecnt + 1.0f);
}
__global__ void k_ggru(const float* __restrict__ shm, const float* __restrict__ ehm,
                       const float* __restrict__ gh,
                       const float* __restrict__ rw, const float* __restrict__ rb,
                       const float* __restrict__ zw, const float* __restrict__ zb,
                       const float* __restrict__ cw, const float* __restrict__ cb,
                       const float* __restrict__ uw, const float* __restrict__ ub,
                       float* __restrict__ ghn) {
  int b = blockIdx.x, tid = threadIdx.x;
  for (int o = tid; o < 512; o += 256) {
    float racc = rb[o], zacc = zb[o];
    for (int k = 0; k < 1536; ++k) {
      float xv = (k < 512) ? shm[b * 512 + k]
               : (k < 1024) ? ehm[b * 512 + k - 512]
                            : gh[b * 512 + k - 1024];
      racc += xv * rw[(size_t)k * 512 + o];
      zacc += xv * zw[(size_t)k * 512 + o];
    }
    float cacc = cb[o], uacc = ub[o];
    for (int k = 0; k < 1024; ++k) {
      float xv = (k < 512) ? shm[b * 512 + k] : ehm[b * 512 + k - 512];
      cacc += xv * cw[(size_t)k * 512 + o];
    }
    for (int k = 0; k < 512; ++k) uacc += gh[b * 512 + k] * uw[(size_t)k * 512 + o];
    float r = sigmoidf_(racc), z = sigmoidf_(zacc);
    float u = tanhf(cacc + r * uacc);
    ghn[b * 512 + o] = z * gh[b * 512 + o] + (1.0f - z) * u;
  }
}
__global__ void k_output(const float* __restrict__ sh, const float* __restrict__ target,
                         const float* __restrict__ gh, float* __restrict__ out) {
  int idx = blockIdx.x * blockDim.x + threadIdx.x;
  if (idx >= 331776) return;
  if (idx < 262144) out[idx] = sh[idx];
  else if (idx < 294912) {
    int t = idx - 262144;
    int j = t & 63, i = (t >> 6) & 63, b = t >> 12;
    out[idx] = target[(b * 64 + i) * 64 + j] + target[(b * 64 + j) * 64 + i];
  } else if (idx < 327680) out[idx] = target[idx - 294912];
  else out[idx] = gh[idx - 327680];
}

// ---------------- host driver ----------------
extern "C" void kernel_launch(void* const* d_in, const int* in_sizes, int n_in,
                              void* d_out, int out_size, void* d_ws, size_t ws_size,
                              hipStream_t stream) {
  const float* sent     = (const float*)d_in[0];
  const float* word     = (const float*)d_in[1];
  const float* target   = (const float*)d_in[2];
  const float* s2sorder = (const float*)d_in[3];
  const int*   matrix   = (const int*)d_in[4];
  const int*   smask    = (const int*)d_in[5];
  const int*   wmask    = (const int*)d_in[6];
  const float* edge_emb = (const float*)d_in[7];
  const float* g1_w = (const float*)d_in[8];  const float* g1_b = (const float*)d_in[9];
  const float* g2_w = (const float*)d_in[10]; const float* g2_b = (const float*)d_in[11];
  const float* s_rw = (const float*)d_in[12]; const float* s_rb = (const float*)d_in[13];
  const float* s_zw = (const float*)d_in[14]; const float* s_zb = (const float*)d_in[15];
  const float* s_cw = (const float*)d_in[16]; const float* s_cb = (const float*)d_in[17];
  const float* s_uw = (const float*)d_in[18]; const float* s_ub = (const float*)d_in[19];
  const float* e_rw = (const float*)d_in[20]; const float* e_rb = (const float*)d_in[21];
  const float* e_zw = (const float*)d_in[22]; const float* e_zb = (const float*)d_in[23];
  const float* e_cw = (const float*)d_in[24]; const float* e_cb = (const float*)d_in[25];
  const float* e_uw = (const float*)d_in[26]; const float* e_ub = (const float*)d_in[27];
  const float* g_rw = (const float*)d_in[28]; const float* g_rb = (const float*)d_in[29];
  const float* g_zw = (const float*)d_in[30]; const float* g_zb = (const float*)d_in[31];
  const float* g_cw = (const float*)d_in[32]; const float* g_cb = (const float*)d_in[33];
  const float* g_uw = (const float*)d_in[34]; const float* g_ub = (const float*)d_in[35];

  char* ws = (char*)d_ws;
  size_t off = 0;
  auto alloc = [&](size_t bytes) -> void* {
    void* p = ws + off;
    off += (bytes + 255) & ~(size_t)255;
    return p;
  };
  auto fragBytes = [](int K, int N) { return (size_t)(K / 32) * (N / 16) * 256 * 4; };

  uint32_t* Fg1 = (uint32_t*)alloc(fragBytes(1152, 640));
  uint32_t* Fg2 = (uint32_t*)alloc(fragBytes(1152, 640));
  uint32_t* Fsr = (uint32_t*)alloc(fragBytes(2688, 512));
  uint32_t* Fsz = (uint32_t*)alloc(fragBytes(2688, 512));
  uint32_t* Fsc = (uint32_t*)alloc(fragBytes(2176, 512));
  uint32_t* Fsu = (uint32_t*)alloc(fragBytes(512, 512));
  uint32_t* Fer = (uint32_t*)alloc(fragBytes(2176, 512));
  uint32_t* Fez = (uint32_t*)alloc(fragBytes(2176, 512));
  uint32_t* Fec = (uint32_t*)alloc(fragBytes(1664, 512));
  uint32_t* Feu = (uint32_t*)alloc(fragBytes(512, 512));
  uint16_t* lab_bf = (uint16_t*)alloc((size_t)Bn * Sn * Wn * Ln * 2);
  float* shA = (float*)alloc(262144 * 4); float* shB = (float*)alloc(262144 * 4);
  float* ehA = (float*)alloc(524288 * 4); float* ehB = (float*)alloc(524288 * 4);
  float* ghA = (float*)alloc(4096 * 4);   float* ghB = (float*)alloc(4096 * 4);
  uint16_t* sh_bf = (uint16_t*)alloc(262144 * 2);
  uint16_t* eh_bf = (uint16_t*)alloc(524288 * 2);
  float* sns = (float*)alloc(262144 * 4);
  float* sne = (float*)alloc((size_t)Bn * Sn * 640 * 4);
  float* ens = (float*)alloc((size_t)Bn * Wn * 640 * 4);
  uint16_t* xh_s = (uint16_t*)alloc((size_t)512 * 2688 * 2);
  uint16_t* xh_e = (uint16_t*)alloc((size_t)1024 * 2176 * 2);
  float* r_s = (float*)alloc(262144 * 4); float* z_s = (float*)alloc(262144 * 4);
  float* c_s = (float*)alloc(262144 * 4); float* ul_s = (float*)alloc(262144 * 4);
  float* r_e = (float*)alloc(524288 * 4); float* z_e = (float*)alloc(524288 * 4);
  float* c_e = (float*)alloc(524288 * 4); float* ul_e = (float*)alloc(524288 * 4);
  float* shm = (float*)alloc(4096 * 4);   float* ehm = (float*)alloc(4096 * 4);
  (void)in_sizes; (void)n_in; (void)out_size; (void)ws_size;

  auto pack = [&](const float* W, uint32_t* F, int K, int N) {
    int total = (K / 32) * (N / 16) * 256;
    k_pack_w<<<(total + 255) / 256, 256, 0, stream>>>(W, F, K, N);
  };
  pack(g1_w, Fg1, 1152, 640);  pack(g2_w, Fg2, 1152, 640);
  pack(s_rw, Fsr, 2688, 512);  pack(s_zw, Fsz, 2688, 512);
  pack(s_cw, Fsc, 2176, 512);  pack(s_uw, Fsu, 512, 512);
  pack(e_rw, Fer, 2176, 512);  pack(e_zw, Fez, 2176, 512);
  pack(e_cw, Fec, 1664, 512);  pack(e_uw, Feu, 512, 512);

  {
    size_t tot = (size_t)Bn * Sn * Wn * Ln;
    k_label_bf<<<(unsigned)((tot + 255) / 256), 256, 0, stream>>>(matrix, edge_emb, lab_bf);
  }
  k_zero<<<(262144 + 255) / 256, 256, 0, stream>>>(shA, 262144);
  k_zero<<<(524288 + 255) / 256, 256, 0, stream>>>(ehA, 524288);
  k_zero<<<(4096 + 255) / 256, 256, 0, stream>>>(ghA, 4096);

  float* sh = shA; float* sh_n = shB;
  float* eh = ehA; float* eh_n = ehB;
  float* gh = ghA; float* gh_n = ghB;

  for (int layer = 0; layer < 2; ++layer) {
    k_f2bf<<<(262144 + 255) / 256, 256, 0, stream>>>(sh, sh_bf, 262144);
    k_f2bf<<<(524288 + 255) / 256, 256, 0, stream>>>(eh, eh_bf, 524288);
    k_sns<<<(262144 + 255) / 256, 256, 0, stream>>>(s2sorder, sh, sns);

    dim3 gg1(Bn * Sn, 10);
    k_gate_s<<<gg1, 256, 0, stream>>>(sh_bf, eh_bf, lab_bf, eh, edge_emb, matrix, Fg1, g1_b, sne);
    dim3 gg2(Bn * Wn, 10);
    k_gate_e<<<gg2, 256, 0, stream>>>(sh_bf, eh_bf, lab_bf, sh, edge_emb, matrix, Fg2, g2_b, ens);

    k_concat_s<<<(512 * 2688 + 255) / 256, 256, 0, stream>>>(sent, sns, sne, gh, sh, xh_s);
    k_concat_e<<<(1024 * 2176 + 255) / 256, 256, 0, stream>>>(word, ens, gh, eh, xh_e);

    dim3 gs(4, 8), ge(8, 8);
    k_gemm_bf16<<<gs, 256, 0, stream>>>(xh_s, 2688, Fsr, 32, s_rb, r_s, 512, 2688);
    k_gemm_bf16<<<gs, 256, 0, stream>>>(xh_s, 2688, Fsz, 32, s_zb, z_s, 512, 2688);
    k_gemm_bf16<<<gs, 256, 0, stream>>>(xh_s, 2688, Fsc, 32, s_cb, c_s, 512, 2176);
    k_gemm_bf16<<<gs, 256, 0, stream>>>(xh_s + 2176, 2688, Fsu, 32, s_ub, ul_s, 512, 512);
    k_gemm_bf16<<<ge, 256, 0, stream>>>(xh_e, 2176, Fer, 32, e_rb, r_e, 512, 2176);
    k_gemm_bf16<<<ge, 256, 0, stream>>>(xh_e, 2176, Fez, 32, e_zb, z_e, 512, 2176);
    k_gemm_bf16<<<ge, 256, 0, stream>>>(xh_e, 2176, Fec, 32, e_cb, c_e, 512, 1664);
    k_gemm_bf16<<<ge, 256, 0, stream>>>(xh_e + 1664, 2176, Feu, 32, e_ub, ul_e, 512, 512);

    k_gru_point<<<(262144 + 255) / 256, 256, 0, stream>>>(r_s, z_s, c_s, ul_s, sh, smask, sh_n, 512);
    k_gru_point<<<(524288 + 255) / 256, 256, 0, stream>>>(r_e, z_e, c_e, ul_e, eh, wmask, eh_n, 1024);
    k_means<<<(4096 + 255) / 256, 256, 0, stream>>>(sh_n, eh_n, smask, wmask, shm, ehm);
    k_ggru<<<Bn, 256, 0, stream>>>(shm, ehm, gh, g_rw, g_rb, g_zw, g_zb, g_cw, g_cb, g_uw, g_ub, gh_n);

    float* t;
    t = sh; sh = sh_n; sh_n = t;
    t = eh; eh = eh_n; eh_n = t;
    t = gh; gh = gh_n; gh_n = t;
  }

  k_output<<<(331776 + 255) / 256, 256, 0, stream>>>(sh, target, gh, (float*)d_out);
}
